// WindowMultiHeadAttention_41257455845721
// MI455X (gfx1250) — matrix operand order repaired and checked by reading, not compiled, not run on hardware
//
#include <hip/hip_runtime.h>

// Window multi-head attention for MI455X (gfx1250), wave32 + WMMA f16.
// B=1024 windows, L=64, C=1024, H=16, d_k=64.

#define WIN   64
#define NH    16
#define DK    64
#define CDIM  1024
#define NB    1024
#define MTOT  (NB * WIN)     // 65536

#define XLD   (CDIM + 8)     // padded LDS row stride (halves) for 1024-wide panels
#define LDT   (DK + 8)       // padded LDS row stride (halves) for 64-wide tiles

typedef _Float16 h4v  __attribute__((ext_vector_type(4)));
typedef _Float16 h8v  __attribute__((ext_vector_type(8)));
typedef _Float16 v16h __attribute__((ext_vector_type(16)));
typedef float    v8f  __attribute__((ext_vector_type(8)));
typedef float    f4v  __attribute__((ext_vector_type(4)));
typedef int      i4v  __attribute__((ext_vector_type(4)));

#define ASG __attribute__((address_space(1)))
#define ASL __attribute__((address_space(3)))

#if __has_builtin(__builtin_amdgcn_global_load_async_to_lds_b128)
#define HAVE_ASYNC 1
#else
#define HAVE_ASYNC 0
#endif

static __device__ __forceinline__ void wait_async_then_barrier() {
#if HAVE_ASYNC
#if __has_builtin(__builtin_amdgcn_s_wait_asynccnt)
  __builtin_amdgcn_s_wait_asynccnt(0);
#else
  asm volatile("s_wait_asynccnt 0x0" ::: "memory");
#endif
#endif
  __syncthreads();
}

// Copy rows x cols f16 (cols % 8 == 0) from global (srcld) into LDS (dstld),
// 16B granules. Uses GLOBAL_LOAD_ASYNC_TO_LDS_B128 when the toolchain has it:
// signature (v4i AS1* src, v4i AS3* dst, imm offset, imm cpol).
static __device__ __forceinline__ void stage_rows_f16(
    _Float16* dst, int dstld, const _Float16* src, int srcld,
    int rows, int cols, int tid, int nthreads) {
  const int gpr = cols >> 3;  // 16B granules per row
  for (int v = tid; v < rows * gpr; v += nthreads) {
    const int r = v / gpr, gc = v - r * gpr;
#if HAVE_ASYNC
    __builtin_amdgcn_global_load_async_to_lds_b128(
        (ASG i4v*)(src + (size_t)r * srcld + gc * 8),
        (ASL i4v*)(dst + (size_t)r * dstld + gc * 8), 0, 0);
#else
    *(h8v*)(dst + (size_t)r * dstld + gc * 8) =
        *(const h8v*)(src + (size_t)r * srcld + gc * 8);
#endif
  }
}

// ---- WMMA fragment loaders (layouts per CDNA5 ISA 7.12.2) -----------------

// A 16x32 f16: lane m = lane&15; halves 0..7 = K[g*8..g*8+7],
// halves 8..15 = K[g*8+16..g*8+23], g = lane>>4. Row-major [16, ld].
static __device__ __forceinline__ v16h frag_a_h(const _Float16* p, int ld, int lane) {
  const int m = lane & 15, g = lane >> 4;
  const _Float16* q = p + (size_t)m * ld + g * 8;
  h8v lo = *(const h8v*)(q);
  h8v hi = *(const h8v*)(q + 16);
  return __builtin_shufflevector(lo, hi, 0,1,2,3,4,5,6,7,8,9,10,11,12,13,14,15);
}

// B 32x16 f16: lane n = lane&15; halves 0..7 = K[g*8..g*8+7],
// halves 8..15 = K[g*8+16..g*8+23], g = lane>>4.
// Row n of a row-major [16, ld] operand supplies B's column n (y = A * W^T).
static __device__ __forceinline__ v16h frag_b_h(const _Float16* p, int ld, int lane) {
  const int n = lane & 15, g = lane >> 4;
  const _Float16* q = p + (size_t)n * ld + g * 8;
  h8v lo = *(const h8v*)(q);
  h8v hi = *(const h8v*)(q + 16);
  return __builtin_shufflevector(lo, hi, 0,1,2,3,4,5,6,7,8,9,10,11,12,13,14,15);
}

#define WMMA_F16(a, b, c) \
  __builtin_amdgcn_wmma_f32_16x16x32_f16(false, (a), false, (b), (short)0, (c), false, false)

// ---- Kernel 0: f32 -> f16 weight conversion -------------------------------

__global__ void wmha_cvt_f16(const float* __restrict__ src,
                             _Float16* __restrict__ dst, int n) {
  int i = blockIdx.x * blockDim.x + threadIdx.x;
  const int stride = gridDim.x * blockDim.x;
  for (; i < n; i += stride) dst[i] = (_Float16)src[i];
}

// ---- Kernel 1: fused QKV projection ---------------------------------------
// 8 waves / 128 rows per WG. The 128x1024 f32 X panel is converted and staged
// once into LDS (CDNA5: 320KB/WGP); each wave then sweeps all N=1024 columns
// of WQ/WK/WV so X HBM traffic is read-once. Q,K stored [b,h,l,d]; V stored
// transposed [b,h,d,l] so the P*V GEMM sees contiguous B fragments.

__global__ void __launch_bounds__(256)
wmha_qkv(const float* __restrict__ X,
         const _Float16* __restrict__ WQ, const _Float16* __restrict__ WK,
         const _Float16* __restrict__ WV,
         _Float16* __restrict__ Qb, _Float16* __restrict__ Kb,
         _Float16* __restrict__ Vtb) {
  extern __shared__ __align__(16) _Float16 Xl[];   // [128][XLD]
  const int tid = threadIdx.x;
  const int lane = tid & 31;
  const int w = tid >> 5;                 // 0..7: 16-row tile
  const int mbase = blockIdx.x * 128;

  // Stage + convert the 128x1024 f32 panel into padded f16 LDS.
  const float* Xg = X + (size_t)mbase * CDIM;
  for (int v = tid; v < 128 * (CDIM / 4); v += 256) {
    const int row = v >> 8;               // 256 float4 per row
    const int c4 = v & 255;
    f4v x = *(const f4v*)(Xg + (size_t)row * CDIM + c4 * 4);
    h4v h;
#pragma unroll
    for (int i = 0; i < 4; ++i) h[i] = (_Float16)x[i];
    *(h4v*)(&Xl[(size_t)row * XLD + c4 * 4]) = h;
  }
  __syncthreads();

  const _Float16* Arow = &Xl[(size_t)(w * 16) * XLD];
  const int g = lane >> 4, nl = lane & 15;

  for (int nb = 0; nb < 16; ++nb) {
    v8f aq[4] = {}, ak[4] = {}, av[4] = {};
    for (int k = 0; k < CDIM; k += 32) {
      v16h a = frag_a_h(Arow + k, XLD, lane);
#pragma unroll
      for (int t = 0; t < 4; ++t) {
        const size_t nrow = (size_t)(nb * 64 + t * 16) * CDIM + k;
        v16h bq = frag_b_h(WQ + nrow, CDIM, lane);
        aq[t] = WMMA_F16(a, bq, aq[t]);
        v16h bk = frag_b_h(WK + nrow, CDIM, lane);
        ak[t] = WMMA_F16(a, bk, ak[t]);
        v16h bv = frag_b_h(WV + nrow, CDIM, lane);
        av[t] = WMMA_F16(a, bv, av[t]);
      }
    }
#pragma unroll
    for (int t = 0; t < 4; ++t) {
#pragma unroll
      for (int r = 0; r < 8; ++r) {
        const int Mrow = mbase + w * 16 + r + 8 * g;
        const int Ncol = nb * 64 + t * 16 + nl;
        const int b = Mrow >> 6, l = Mrow & 63;
        const int h = Ncol >> 6, d = Ncol & 63;
        const size_t qoff = ((size_t)(b * NH + h) * WIN + l) * DK + d;
        Qb[qoff] = (_Float16)aq[t][r];
        Kb[qoff] = (_Float16)ak[t][r];
        const size_t voff = ((size_t)(b * NH + h) * DK + d) * WIN + l;
        Vtb[voff] = (_Float16)av[t][r];
      }
    }
  }
}

// ---- Kernel 2: attention per (b,h) ----------------------------------------
// Q/K/Vt tiles async-staged into padded LDS; S = Q*K^T via WMMA, bias+mask,
// register softmax across 16-lane halves, attn -> d_out, P via LDS,
// context = P*V via WMMA -> f16 [m, h*64+d].

__global__ void __launch_bounds__(128)
wmha_attn(const _Float16* __restrict__ Qb, const _Float16* __restrict__ Kb,
          const _Float16* __restrict__ Vtb, const float* __restrict__ mask,
          const float* __restrict__ bias_table,
          float* __restrict__ attn_out, _Float16* __restrict__ Ctx) {
  __shared__ __align__(16) _Float16 Qs[WIN * LDT];
  __shared__ __align__(16) _Float16 Ks[WIN * LDT];
  __shared__ __align__(16) _Float16 Vs[WIN * LDT];
  __shared__ __align__(16) _Float16 Ps[WIN * LDT];

  const int tid = threadIdx.x;
  const int lane = tid & 31;
  const int w = tid >> 5;               // 0..3: 16-row tile
  const int bid = blockIdx.x;           // b*NH + h
  const int b = bid >> 4, h = bid & 15;

  stage_rows_f16(Qs, LDT, Qb + (size_t)bid * WIN * DK, DK, WIN, DK, tid, 128);
  stage_rows_f16(Ks, LDT, Kb + (size_t)bid * WIN * DK, DK, WIN, DK, tid, 128);
  stage_rows_f16(Vs, LDT, Vtb + (size_t)bid * DK * WIN, WIN, DK, WIN, tid, 128);
  wait_async_then_barrier();

  // ---- S = Q K^T ----
  v8f s[4] = {};
  for (int k = 0; k < DK; k += 32) {
    v16h a = frag_a_h(&Qs[(size_t)(w * 16) * LDT + k], LDT, lane);
#pragma unroll
    for (int t = 0; t < 4; ++t) {
      v16h bf = frag_b_h(&Ks[(size_t)(t * 16) * LDT + k], LDT, lane);
      s[t] = WMMA_F16(a, bf, s[t]);
    }
  }

  const int g = lane >> 4, nl = lane & 15;
  const float* mrow = mask + (size_t)(b & 63) * WIN * WIN;

  // ---- scale + bias + mask, row softmax, emit attn & P ----
#pragma unroll
  for (int r = 0; r < 8; ++r) {
    const int l = w * 16 + r + 8 * g;
    float v[4];
    float mx = -1e30f;
#pragma unroll
    for (int t = 0; t < 4; ++t) {
      const int m = t * 16 + nl;
      float x = s[t][r] * 0.125f
              + bias_table[(l - m + WIN - 1) * NH + h]
              + mrow[l * WIN + m];
      v[t] = x;
      mx = fmaxf(mx, x);
    }
    for (int off = 8; off; off >>= 1) mx = fmaxf(mx, __shfl_xor(mx, off, 16));
    float sum = 0.f;
#pragma unroll
    for (int t = 0; t < 4; ++t) { v[t] = __expf(v[t] - mx); sum += v[t]; }
    for (int off = 8; off; off >>= 1) sum += __shfl_xor(sum, off, 16);
    const float inv = 1.0f / sum;
#pragma unroll
    for (int t = 0; t < 4; ++t) {
      const int m = t * 16 + nl;
      const float p = v[t] * inv;
      attn_out[(size_t)bid * WIN * WIN + l * WIN + m] = p;
      Ps[l * LDT + m] = (_Float16)p;
    }
  }
  __syncthreads();

  // ---- context = P * V (B from transposed V tile in LDS) ----
  v8f c[4] = {};
  for (int k = 0; k < WIN; k += 32) {
    v16h a = frag_a_h(&Ps[(size_t)(w * 16) * LDT + k], LDT, lane);
#pragma unroll
    for (int t = 0; t < 4; ++t) {
      v16h bf = frag_b_h(&Vs[(size_t)(t * 16) * LDT + k], LDT, lane);
      c[t] = WMMA_F16(a, bf, c[t]);
    }
  }
#pragma unroll
  for (int t = 0; t < 4; ++t) {
#pragma unroll
    for (int r = 0; r < 8; ++r) {
      const int l = w * 16 + r + 8 * g;
      const int d = t * 16 + nl;
      Ctx[((size_t)(b * WIN + l)) * CDIM + h * DK + d] = (_Float16)c[t][r];
    }
  }
}

// ---- Kernel 3: output projection ------------------------------------------
// Same read-once LDS panel scheme as QKV; Ctx panel is already f16 so it is
// staged with async LDS copies when available.

__global__ void __launch_bounds__(256)
wmha_proj(const _Float16* __restrict__ Ctx, const _Float16* __restrict__ Wfc,
          float* __restrict__ out) {
  extern __shared__ __align__(16) _Float16 Cl[];   // [128][XLD]
  const int tid = threadIdx.x;
  const int lane = tid & 31;
  const int w = tid >> 5;                 // 0..7
  const int mbase = blockIdx.x * 128;

  stage_rows_f16(Cl, XLD, Ctx + (size_t)mbase * CDIM, CDIM, 128, CDIM, tid, 256);
  wait_async_then_barrier();

  const _Float16* Arow = &Cl[(size_t)(w * 16) * XLD];
  const int g = lane >> 4, nl = lane & 15;

  for (int nb = 0; nb < 16; ++nb) {
    v8f acc[4] = {};
    for (int k = 0; k < CDIM; k += 32) {
      v16h a = frag_a_h(Arow + k, XLD, lane);
#pragma unroll
      for (int t = 0; t < 4; ++t) {
        v16h bf = frag_b_h(Wfc + (size_t)(nb * 64 + t * 16) * CDIM + k, CDIM, lane);
        acc[t] = WMMA_F16(a, bf, acc[t]);
      }
    }
#pragma unroll
    for (int t = 0; t < 4; ++t) {
#pragma unroll
      for (int r = 0; r < 8; ++r) {
        const int Mrow = mbase + w * 16 + r + 8 * g;
        const int Ncol = nb * 64 + t * 16 + nl;
        out[(size_t)Mrow * CDIM + Ncol] = acc[t][r];
      }
    }
  }
}

// ---- Host launch -----------------------------------------------------------

extern "C" void kernel_launch(void* const* d_in, const int* in_sizes, int n_in,
                              void* d_out, int out_size, void* d_ws, size_t ws_size,
                              hipStream_t stream) {
  (void)in_sizes; (void)n_in; (void)out_size; (void)ws_size;
  const float* X    = (const float*)d_in[0];  // [1024, 64, 1024]
  const float* mask = (const float*)d_in[1];  // [64, 64, 64]
  const float* WQ   = (const float*)d_in[2];  // [1024, 1024]
  const float* WK   = (const float*)d_in[3];
  const float* WV   = (const float*)d_in[4];
  const float* WFC  = (const float*)d_in[5];
  const float* bias = (const float*)d_in[6];  // [127, 16]

  float* out = (float*)d_out;                      // [65536, 1024]
  float* attn_out = out + (size_t)MTOT * CDIM;     // [1024, 16, 64, 64]

  const size_t wsz   = (size_t)CDIM * CDIM;        // 1M halves per weight
  const size_t qkvsz = (size_t)NB * NH * WIN * DK; // 67.1M halves

  _Float16* WQh  = (_Float16*)d_ws;
  _Float16* WKh  = WQh + wsz;
  _Float16* WVh  = WKh + wsz;
  _Float16* WFCh = WVh + wsz;
  _Float16* Qb   = WFCh + wsz;
  _Float16* Kb   = Qb + qkvsz;
  _Float16* Vtb  = Kb + qkvsz;
  _Float16* Ctx  = Vtb + qkvsz;

  const size_t panel_bytes = (size_t)128 * XLD * sizeof(_Float16);  // ~258KB

  wmha_cvt_f16<<<1024, 256, 0, stream>>>(WQ,  WQh,  (int)wsz);
  wmha_cvt_f16<<<1024, 256, 0, stream>>>(WK,  WKh,  (int)wsz);
  wmha_cvt_f16<<<1024, 256, 0, stream>>>(WV,  WVh,  (int)wsz);
  wmha_cvt_f16<<<1024, 256, 0, stream>>>(WFC, WFCh, (int)wsz);

  wmha_qkv<<<MTOT / 128, 256, panel_bytes, stream>>>(X, WQh, WKh, WVh,
                                                     Qb, Kb, Vtb);
  wmha_attn<<<NB * NH, 128, 0, stream>>>(Qb, Kb, Vtb, mask, bias,
                                         attn_out, Ctx);
  wmha_proj<<<MTOT / 128, 256, panel_bytes, stream>>>(Ctx, WFCh, out);
}
